// LVDecisionModule_36661840838746
// MI455X (gfx1250) — matrix operand order repaired and checked by reading, not compiled, not run on hardware
//
#include <hip/hip_runtime.h>

typedef __attribute__((ext_vector_type(16))) __bf16 bf16x16;
typedef __attribute__((ext_vector_type(8)))  float  floatx8;

#define BB 256
#define NV 196
#define LT 32
#define CV 768
#define CT 512
#define CI 768
#define NH 12
#define DH 64
#define MQ (BB * NV)   // 50176 rows of q / fused
#define MT (BB * LT)   // 8192 rows of k / v

// ---------------- WMMA + async helpers (CDNA5 wave32) ----------------

__device__ __forceinline__ floatx8 wmma_bf16(bf16x16 a, bf16x16 b, floatx8 c) {
  return __builtin_amdgcn_wmma_f32_16x16x32_bf16(
      /*neg_a=*/false, a, /*neg_b=*/false, b,
      /*c_mod=*/(short)0, c, /*reuse_a=*/false, /*reuse_b=*/false);
}

// Async Global->LDS 16B copy (tracked by ASYNCcnt). LDS dest address is the
// low 32 bits of the generic pointer (LDS aperture: addr[31:0] == offset).
__device__ __forceinline__ void async_copy_b128(void* lds_dst, const void* gsrc) {
  unsigned dst = (unsigned)(size_t)lds_dst;
  unsigned long long src = (unsigned long long)(size_t)gsrc;
  asm volatile("global_load_async_to_lds_b128 %0, %1, off"
               :: "v"(dst), "v"(src) : "memory");
}
__device__ __forceinline__ void wait_async0() {
  asm volatile("s_wait_asynccnt 0" ::: "memory");
}

// A-matrix 16x32 bf16 fragment. p -> (row, kbase), kbase = k0 + 16*(lane>=16).
// Elements 0..15 = K kbase..kbase+15 (contiguous, matches B fragments).
__device__ __forceinline__ bf16x16 frag_a_contig(const __bf16* p) {
  bf16x16 f;
#pragma unroll
  for (int i = 0; i < 8; ++i) { f[i] = p[i]; f[8 + i] = p[8 + i]; }
  return f;
}

// B-matrix 32x16 bf16 fragment, 16 contiguous elements at p (natural K-major
// or fragment-packed weight layout).
__device__ __forceinline__ bf16x16 frag_b_contig(const __bf16* p) {
  bf16x16 f;
#pragma unroll
  for (int e = 0; e < 16; ++e) f[e] = p[e];
  return f;
}

// ---------------- elementwise kernels ----------------

__global__ void cvt_f32_bf16_kernel(const float* __restrict__ in,
                                    __bf16* __restrict__ out, int n) {
  int i = blockIdx.x * blockDim.x + threadIdx.x;
  if (i < n) out[i] = (__bf16)in[i];
}

__global__ void fill_f32_kernel(float* __restrict__ p, float v, int n) {
  int i = blockIdx.x * blockDim.x + threadIdx.x;
  if (i < n) p[i] = v;
}

// Pack f32 weight W[K,N] into WMMA B-fragment order (bf16):
// P[(nt*(K/32) + kb)*512 + lane*16 + e] = W[(kb*32 + (lane>>4)*16 + e)*N + nt*16 + (lane&15)]
__global__ void pack_w_bf16_kernel(const float* __restrict__ W,
                                   __bf16* __restrict__ P, int K, int N) {
  int idx = blockIdx.x * blockDim.x + threadIdx.x;
  if (idx >= K * N) return;
  const int e = idx & 15;
  const int lane = (idx >> 4) & 31;
  const int blk = idx >> 9;
  const int kblocks = K >> 5;
  const int kb = blk % kblocks;
  const int nt = blk / kblocks;
  const int k = kb * 32 + (lane >> 4) * 16 + e;
  const int col = nt * 16 + (lane & 15);
  P[idx] = (__bf16)W[(size_t)k * N + col];
}

// ---------------- GEMM: O = A[M,K](bf16) @ Wpacked + bias ----------------
// block = 256 threads = 8 waves (2 row x 4 col); wave tile = 16x64 (4 WMMAs
// per A fragment); block tile = 32x256. A tile async-staged in LDS (padded
// rows: stride K+8 -> conflict-free b128 reads). TRANS_V -> [M/32, N, 32].

template <int K, bool TRANS_V>
__global__ __launch_bounds__(256) void gemm_bias_bf16_kernel(
    const __bf16* __restrict__ A, const __bf16* __restrict__ Wp,
    const float* __restrict__ bias, __bf16* __restrict__ O, int N) {
  constexpr int LDA = K + 8;          // padded LDS row stride
  constexpr int KCH = K / 8;          // 16B chunks per row
  __shared__ __bf16 lds_a[32 * LDA];

  const int wv   = threadIdx.x >> 5;
  const int lane = threadIdx.x & 31;
  const int lo = lane & 15, hi = lane >> 4;
  const int row0 = blockIdx.x * 32 + (wv & 1) * 16;
  const int col0 = blockIdx.y * 256 + (wv >> 1) * 64;
  constexpr int kblocks = K >> 5;

  // cooperative async copy of the (contiguous) 32xK A tile into LDS
  {
    const __bf16* asrc = A + (size_t)blockIdx.x * 32 * K;
#pragma unroll
    for (int it = 0; it < K / 64; ++it) {
      const int ci = it * 256 + threadIdx.x;       // chunk id
      const int r = ci / KCH, c = (ci % KCH) * 8;
      async_copy_b128(&lds_a[r * LDA + c], asrc + (size_t)r * K + c);
    }
    wait_async0();
    __syncthreads();
  }

  floatx8 acc[4] = {};
  const __bf16* ap = lds_a + (size_t)((wv & 1) * 16 + lo) * LDA + hi * 16;
  const __bf16* bp = Wp + ((size_t)(col0 >> 4) * kblocks) * 512 + lane * 16;
  constexpr size_t tstride = (size_t)kblocks * 512;

  for (int k0 = 0; k0 < K; k0 += 32) {
    bf16x16 af = frag_a_contig(ap + k0);
#pragma unroll
    for (int q4 = 0; q4 < 4; ++q4)
      acc[q4] = wmma_bf16(af, frag_b_contig(bp + q4 * tstride), acc[q4]);
    bp += 512;
  }

#pragma unroll
  for (int q4 = 0; q4 < 4; ++q4) {
    const int col = col0 + q4 * 16 + lo;
    const float bvv = bias[col];
#pragma unroll
    for (int i = 0; i < 8; ++i) {
      const int row = row0 + hi * 8 + i;
      const float v = acc[q4][i] + bvv;
      if (TRANS_V)  // [row/32, col, row%32]: K contiguous for ctx B-fragments
        O[((size_t)(row >> 5) * N + col) * 32 + (row & 31)] = (__bf16)v;
      else
        O[(size_t)row * N + col] = (__bf16)v;
    }
  }
}

// ---------------- attention: scores->softmax->ctx->fused ----------------
// block = 128 threads = 4 waves; block = one (batch, 16-row Nv tile);
// each wave handles 3 heads. rel output is analytically 1/32.

#define SCS 33   // padded f32 score row stride (bank-conflict-free)
#define ATS 40   // padded bf16 attn row stride

__global__ __launch_bounds__(128) void attn_kernel(
    const __bf16* __restrict__ qb, const __bf16* __restrict__ kb,
    const __bf16* __restrict__ vt,   // [B, CI, LT] transposed v
    const float* __restrict__ vis, __bf16* __restrict__ fused) {
  __shared__ float  sc[4][16 * SCS];
  __shared__ __bf16 at[4][16 * ATS];

  const int w    = threadIdx.x >> 5;
  const int lane = threadIdx.x & 31;
  const int lo = lane & 15, hi = lane >> 4;
  const int b  = blockIdx.x / 13;
  const int t  = blockIdx.x % 13;
  const int m0 = t * 16;

  int mr = m0 + lo; if (mr > NV - 1) mr = NV - 1;   // clamp ragged tile reads
  const int arow = b * NV + mr;

  for (int hh = 0; hh < 3; ++hh) {
    const int h = w + hh * 4;
    const int hoff = h * DH;

    // ---- scores = (q @ k^T) / 8 : M=16, N=32 (two tiles), K=64
    floatx8 s0 = {}, s1 = {};
#pragma unroll
    for (int k0 = 0; k0 < DH; k0 += 32) {
      bf16x16 af = frag_a_contig(qb + (size_t)arow * CI + hoff + k0 + hi * 16);
      bf16x16 b0 = frag_b_contig(kb + (size_t)(b * LT + lo) * CI + hoff + k0 + hi * 16);
      bf16x16 b1 = frag_b_contig(kb + (size_t)(b * LT + 16 + lo) * CI + hoff + k0 + hi * 16);
      s0 = wmma_bf16(af, b0, s0);
      s1 = wmma_bf16(af, b1, s1);
    }
#pragma unroll
    for (int i = 0; i < 8; ++i) {
      sc[w][(hi * 8 + i) * SCS + lo]      = s0[i] * 0.125f;
      sc[w][(hi * 8 + i) * SCS + 16 + lo] = s1[i] * 0.125f;
    }
    asm volatile("s_wait_dscnt 0" ::: "memory");   // same-wave LDS ordering

    // ---- softmax over Lt=32 per row; lanes 0..15 each own one row
    if (lane < 16) {
      const float* row = &sc[w][lane * SCS];
      float mx = row[0];
#pragma unroll
      for (int j = 1; j < 32; ++j) mx = fmaxf(mx, row[j]);
      float e[32]; float sum = 0.f;
#pragma unroll
      for (int j = 0; j < 32; ++j) { e[j] = __expf(row[j] - mx); sum += e[j]; }
      const float inv = 1.f / sum;
#pragma unroll
      for (int j = 0; j < 32; ++j) at[w][lane * ATS + j] = (__bf16)(e[j] * inv);
    }
    asm volatile("s_wait_dscnt 0" ::: "memory");

    // ---- ctx = attn @ v : M=16, K=32, N=64 (4 tiles); fused = ctx + vis
    bf16x16 afr = frag_a_contig(&at[w][lo * ATS + hi * 16]);
#pragma unroll
    for (int dt = 0; dt < 4; ++dt) {
      const int col = hoff + dt * 16 + lo;
      floatx8 c = {};
      bf16x16 bfr = frag_b_contig(vt + ((size_t)b * CI + col) * LT + hi * 16);
      c = wmma_bf16(afr, bfr, c);
#pragma unroll
      for (int i = 0; i < 8; ++i) {
        const int m = m0 + hi * 8 + i;
        if (m < NV) {
          const size_t r = (size_t)(b * NV + m) * CI + col;
          fused[r] = (__bf16)(c[i] + vis[r]);
        }
      }
    }
  }
}

// ---------------- MLP: logits = gelu(fused@W1+b1)@W2+b2 ----------------
// block = 256 threads = 8 waves; block owns a 16-row tile (async-staged in
// LDS, shared by all waves); waves sweep the 48 N-tiles of packed W1 (K=768),
// quickGELU, contract with tiny W2 via LDS atomics. h never hits global.

__global__ __launch_bounds__(256) void mlp_kernel(
    const __bf16* __restrict__ fused, const __bf16* __restrict__ w1p,
    const float* __restrict__ b1, const float* __restrict__ w2,
    const float* __restrict__ b2, float* __restrict__ out) {
  constexpr int LDF = CI + 8;
  __shared__ __bf16 lds_f[16 * LDF];
  __shared__ float ll[32];

  const int w    = threadIdx.x >> 5;
  const int lane = threadIdx.x & 31;
  const int lo = lane & 15, hi = lane >> 4;
  const int row0 = blockIdx.x * 16;
  constexpr int kblocks = CI >> 5;   // 24

  if (threadIdx.x < 32) ll[threadIdx.x] = 0.f;
  {
    const __bf16* fsrc = fused + (size_t)row0 * CI;
#pragma unroll
    for (int it = 0; it < 6; ++it) {       // 16*768/(256*8) = 6 chunks/thread
      const int ci = it * 256 + threadIdx.x;
      const int r = ci / (CI / 8), c = (ci % (CI / 8)) * 8;
      async_copy_b128(&lds_f[r * LDF + c], fsrc + (size_t)r * CI + c);
    }
    wait_async0();
    __syncthreads();
  }

  float l0[8], l1[8];
#pragma unroll
  for (int i = 0; i < 8; ++i) { l0[i] = 0.f; l1[i] = 0.f; }

  const __bf16* ap = lds_f + (size_t)lo * LDF + hi * 16;
  for (int nt = w; nt < 48; nt += 8) {
    const __bf16* bp = w1p + (size_t)nt * kblocks * 512 + lane * 16;
    floatx8 acc = {};
#pragma unroll 4
    for (int k0 = 0; k0 < CI; k0 += 32) {
      acc = wmma_bf16(frag_a_contig(ap + k0), frag_b_contig(bp), acc);
      bp += 512;
    }
    const int n = nt * 16 + lo;
    const float bias = b1[n];
    const float w2a = w2[n * 2 + 0], w2b = w2[n * 2 + 1];
#pragma unroll
    for (int i = 0; i < 8; ++i) {
      const float x = acc[i] + bias;
      const float g = x / (1.f + __expf(-1.702f * x));   // quick_gelu
      l0[i] += g * w2a;
      l1[i] += g * w2b;
    }
  }
#pragma unroll
  for (int i = 0; i < 8; ++i) {
    atomicAdd(&ll[(hi * 8 + i) * 2 + 0], l0[i]);
    atomicAdd(&ll[(hi * 8 + i) * 2 + 1], l1[i]);
  }
  __syncthreads();
  if (threadIdx.x < 32) {
    const int m = threadIdx.x >> 1, j = threadIdx.x & 1;
    out[(size_t)(row0 + m) * 2 + j] = ll[threadIdx.x] + b2[j];
  }
}

// ---------------- launch ----------------

extern "C" void kernel_launch(void* const* d_in, const int* in_sizes, int n_in,
                              void* d_out, int out_size, void* d_ws, size_t ws_size,
                              hipStream_t stream) {
  (void)in_sizes; (void)n_in; (void)out_size; (void)ws_size;
  const float* vis  = (const float*)d_in[0];
  const float* text = (const float*)d_in[1];
  const float* Wq = (const float*)d_in[2];  const float* bq = (const float*)d_in[3];
  const float* Wk = (const float*)d_in[4];  const float* bk = (const float*)d_in[5];
  const float* Wv = (const float*)d_in[6];  const float* bv = (const float*)d_in[7];
  const float* W1 = (const float*)d_in[8];  const float* b1 = (const float*)d_in[9];
  const float* W2 = (const float*)d_in[10]; const float* b2 = (const float*)d_in[11];
  float* out = (float*)d_out;

  // workspace layout (bf16 elements)
  __bf16* ws = (__bf16*)d_ws;
  const size_t n_vis  = (size_t)MQ * CV;
  const size_t n_text = (size_t)MT * CT;
  const size_t n_wq = (size_t)CV * CI, n_wk = (size_t)CT * CI;
  const size_t n_wv = (size_t)CT * CI, n_w1 = (size_t)CI * CI;
  const size_t n_q = (size_t)MQ * CI, n_k = (size_t)MT * CI, n_v = (size_t)MT * CI;

  __bf16* vis_b  = ws;                 size_t off = n_vis;
  __bf16* text_b = ws + off;           off += n_text;
  __bf16* wq_p   = ws + off;           off += n_wq;
  __bf16* wk_p   = ws + off;           off += n_wk;
  __bf16* wv_p   = ws + off;           off += n_wv;
  __bf16* w1_p   = ws + off;           off += n_w1;
  __bf16* q_b    = ws + off;           off += n_q;
  __bf16* k_b    = ws + off;           off += n_k;
  __bf16* v_t    = ws + off;           off += n_v;
  __bf16* fused_b = ws + off;

  // activations: plain f32->bf16 convert
  cvt_f32_bf16_kernel<<<(unsigned)((n_vis + 255) / 256), 256, 0, stream>>>(vis, vis_b, (int)n_vis);
  cvt_f32_bf16_kernel<<<(unsigned)((n_text + 255) / 256), 256, 0, stream>>>(text, text_b, (int)n_text);

  // weights: convert + pack into WMMA B-fragment order
  pack_w_bf16_kernel<<<(unsigned)((n_wq + 255) / 256), 256, 0, stream>>>(Wq, wq_p, CV, CI);
  pack_w_bf16_kernel<<<(unsigned)((n_wk + 255) / 256), 256, 0, stream>>>(Wk, wk_p, CT, CI);
  pack_w_bf16_kernel<<<(unsigned)((n_wv + 255) / 256), 256, 0, stream>>>(Wv, wv_p, CT, CI);
  pack_w_bf16_kernel<<<(unsigned)((n_w1 + 255) / 256), 256, 0, stream>>>(W1, w1_p, CI, CI);

  // q/k/v projections (bf16 WMMA, f32 accumulate); v stored transposed
  gemm_bias_bf16_kernel<CV, false><<<dim3(MQ / 32, CI / 256), 256, 0, stream>>>(
      vis_b, wq_p, bq, q_b, CI);
  gemm_bias_bf16_kernel<CT, false><<<dim3(MT / 32, CI / 256), 256, 0, stream>>>(
      text_b, wk_p, bk, k_b, CI);
  gemm_bias_bf16_kernel<CT, true><<<dim3(MT / 32, CI / 256), 256, 0, stream>>>(
      text_b, wv_p, bv, v_t, CI);

  // attention + residual fuse
  attn_kernel<<<BB * 13, 128, 0, stream>>>(q_b, k_b, v_t, vis, fused_b);

  // MLP head -> logits (out[0 .. MQ*2))
  mlp_kernel<<<MQ / 16, 256, 0, stream>>>(fused_b, w1_p, b1, W2, b2, out);

  // rel_BN = attn.mean(Lt).mean(H) == 1/32 exactly (softmax rows sum to 1)
  fill_f32_kernel<<<(MQ + 255) / 256, 256, 0, stream>>>(out + (size_t)MQ * 2,
                                                        1.0f / 32.0f, MQ);
}
